// GATReg_33406255628687
// MI455X (gfx1250) — compile-verified, matrix operand order checked by reading, not run on hardware
//
#include <hip/hip_runtime.h>
#include <cstdint>
#include <cstddef>

typedef __attribute__((ext_vector_type(16))) _Float16 v16h;
typedef __attribute__((ext_vector_type(8)))  float    v8f;

#define LEAKY 0.2f
#define ENC_NEG_INF 0x007FFFFFu   // enc(-inf) = ~0xFF800000

// ---- monotonic float <-> uint encoding for atomic max on floats ----
__device__ __forceinline__ unsigned enc_f32(float f) {
  unsigned u = __float_as_uint(f);
  return (u & 0x80000000u) ? ~u : (u | 0x80000000u);
}
__device__ __forceinline__ float dec_f32(unsigned e) {
  unsigned u = (e & 0x80000000u) ? (e ^ 0x80000000u) : ~e;
  return __uint_as_float(u);
}

// =====================================================================
// Kernel 0: initialize accumulators (+ seed output with bias2)
// =====================================================================
__global__ void init_ws(unsigned* __restrict__ m1, float* __restrict__ den1,
                        float* __restrict__ agg1, unsigned* __restrict__ m2,
                        float* __restrict__ den2, float* __restrict__ out,
                        const float* __restrict__ bias2, int N) {
  long total = (long)N * 256;
  long stride = (long)gridDim.x * blockDim.x;
  for (long i = (long)blockIdx.x * blockDim.x + threadIdx.x; i < total; i += stride) {
    agg1[i] = 0.f;
    if (i < (long)N * 4) { m1[i] = ENC_NEG_INF; den1[i] = 0.f; }
    if (i < N)           { m2[i] = ENC_NEG_INF; den2[i] = 0.f; out[i] = bias2[0]; }
  }
}

// =====================================================================
// Kernel 1: h = x @ W1  via v_wmma_f32_16x16x32_f16
//   x: [N,128] f32 (cast to f16 on load), W1: [128,256] f32, h: [N,256] f32
//   One wave computes one 16x16 output tile; K=128 in 4 steps of 32.
//   A layout (16-bit 16x32): lanes 0-15 M=0..15 K=0..7,16..23;
//                            lanes 16-31 K=8..15,24..31 (per ISA 7.12.2).
// =====================================================================
__global__ __launch_bounds__(256) void gemm_wmma(
    const float* __restrict__ x, const float* __restrict__ W,
    float* __restrict__ h, int N) {
  const int wid  = (blockIdx.x * blockDim.x + threadIdx.x) >> 5;
  const int lane = threadIdx.x & 31;
  const int tile_m = wid >> 4;       // 256/16 = 16 tiles along output cols
  const int tile_n = wid & 15;
  if (tile_m * 16 >= N) return;      // uniform across wave
  const int m0 = tile_m * 16, n0 = tile_n * 16;
  const int half = lane >> 4, r = lane & 15;
  int row = m0 + r; if (row >= N) row = N - 1;   // keep EXEC all-ones for WMMA
  const float* xrow = x + (size_t)row * 128;

  v8f acc = {};
  #pragma unroll
  for (int k0 = 0; k0 < 128; k0 += 32) {
    v16h a, b;
    #pragma unroll
    for (int i = 0; i < 8; ++i) {
      const int kp = (i < 4) ? (half * 8 + 2 * i) : (16 + half * 8 + 2 * (i - 4));
      a[2*i]   = (_Float16)xrow[k0 + kp];
      a[2*i+1] = (_Float16)xrow[k0 + kp + 1];
      b[2*i]   = (_Float16)W[(size_t)(k0 + kp)     * 256 + n0 + r];
      b[2*i+1] = (_Float16)W[(size_t)(k0 + kp + 1) * 256 + n0 + r];
    }
    acc = __builtin_amdgcn_wmma_f32_16x16x32_f16(
        /*neg_a=*/false, a, /*neg_b=*/false, b,
        /*c_mod=*/(short)0, acc, /*reuse_a=*/false, /*reuse_b=*/false);
  }
  #pragma unroll
  for (int v = 0; v < 8; ++v) {      // C/D: VGPR v -> M = v (+8 for lanes 16-31)
    const int rr = m0 + half * 8 + v;
    if (rr < N) h[(size_t)rr * 256 + n0 + r] = acc[v];
  }
}

// =====================================================================
// Kernel 2: a_src[n,h] / a_dst[n,h] = <h[n,h,:], att_{src,dst}[h,:]>
//   One wave per node; channel c = j*32+lane stays inside head j/2.
// =====================================================================
__global__ __launch_bounds__(256) void att_reduce(
    const float* __restrict__ h, const float* __restrict__ att_s,
    const float* __restrict__ att_d, float* __restrict__ asrc,
    float* __restrict__ adst, int N) {
  const int wid  = (blockIdx.x * blockDim.x + threadIdx.x) >> 5;
  const int lane = threadIdx.x & 31;
  if (wid >= N) return;
  const float* row = h + (size_t)wid * 256;
  float ps[8], pd[8];
  #pragma unroll
  for (int j = 0; j < 8; ++j) {
    const int c = j * 32 + lane;
    const float v = row[c];
    ps[j] = v * att_s[c];
    pd[j] = v * att_d[c];
  }
  #pragma unroll
  for (int hd = 0; hd < 4; ++hd) {
    float s = ps[2*hd] + ps[2*hd+1];
    float d = pd[2*hd] + pd[2*hd+1];
    #pragma unroll
    for (int off = 16; off > 0; off >>= 1) {
      s += __shfl_xor(s, off, 32);
      d += __shfl_xor(d, off, 32);
    }
    if (lane == 0) { asrc[wid*4 + hd] = s; adst[wid*4 + hd] = d; }
  }
}

// =====================================================================
// Layer-1 edge pass A: e = leaky_relu(a_src[src]+a_dst[dst]); seg-max
// =====================================================================
__global__ void l1_edge_logits(const int* __restrict__ ei,
                               const float* __restrict__ asrc,
                               const float* __restrict__ adst,
                               float* __restrict__ e1,
                               unsigned* __restrict__ m1, int E, int Et) {
  const int gid = blockIdx.x * blockDim.x + threadIdx.x;
  if (gid >= Et * 4) return;
  const int e = gid >> 2, hd = gid & 3;
  int s, d;
  if (e < E) { s = ei[e]; d = ei[E + e]; } else { s = d = e - E; }
  float v = asrc[s*4 + hd] + adst[d*4 + hd];
  v = v > 0.f ? v : LEAKY * v;
  e1[gid] = v;
  atomicMax(&m1[d*4 + hd], enc_f32(v));
}

// Layer-1 edge pass B: w = exp(e - m[dst]); seg-sum
__global__ void l1_edge_exp(const int* __restrict__ ei,
                            float* __restrict__ e1,
                            const unsigned* __restrict__ m1,
                            float* __restrict__ den1, int E, int Et) {
  const int gid = blockIdx.x * blockDim.x + threadIdx.x;
  if (gid >= Et * 4) return;
  const int e = gid >> 2, hd = gid & 3;
  int d;
  if (e < E) { d = ei[E + e]; } else { d = e - E; }
  const float w = __expf(e1[gid] - dec_f32(m1[d*4 + hd]));
  e1[gid] = w;
  atomicAdd(&den1[d*4 + hd], w);
}

// =====================================================================
// Layer-1 edge pass C (heavy): agg[dst] += alpha * h[src]
//   64 threads per edge, float4 per thread -> fully coalesced 1KB gather
//   + contiguous global_atomic_add_f32 scatter.
// =====================================================================
__global__ __launch_bounds__(256) void l1_edge_agg(
    const int* __restrict__ ei, const float* __restrict__ e1,
    const float* __restrict__ den1, const float* __restrict__ h,
    float* __restrict__ agg1, int E, int Et) {
  const int idx = blockIdx.x * 4 + (threadIdx.x >> 6);   // edge id
  if (idx >= Et) return;
  const int sub = threadIdx.x & 63;                      // 64 float4 lanes
  const int hd  = sub >> 4;                              // head = (sub*4)/64
  int s, d;
  if (idx < E) { s = ei[idx]; d = ei[E + idx]; } else { s = d = idx - E; }
  const float alpha = e1[idx*4 + hd] / den1[d*4 + hd];
  const float4 hv = *reinterpret_cast<const float4*>(h + (size_t)s * 256 + sub * 4);
  float* base = agg1 + (size_t)d * 256 + sub * 4;
  atomicAdd(base + 0, alpha * hv.x);
  atomicAdd(base + 1, alpha * hv.y);
  atomicAdd(base + 2, alpha * hv.z);
  atomicAdd(base + 3, alpha * hv.w);
}

// =====================================================================
// Fused: h1 = elu(agg1 + bias1);  h2[n] = <h1[n,:], W2[:,0]>
//   One wave per node, shuffle reduction.
// =====================================================================
__global__ __launch_bounds__(256) void finalize_matvec(
    const float* __restrict__ agg1, const float* __restrict__ bias1,
    const float* __restrict__ W2, float* __restrict__ h2, int N) {
  const int wid  = (blockIdx.x * blockDim.x + threadIdx.x) >> 5;
  const int lane = threadIdx.x & 31;
  if (wid >= N) return;
  float acc = 0.f;
  #pragma unroll
  for (int j = 0; j < 8; ++j) {
    const int c = j * 32 + lane;
    float v = agg1[(size_t)wid * 256 + c] + bias1[c];
    v = v > 0.f ? v : (__expf(v) - 1.f);   // elu, alpha=1
    acc += v * W2[c];
  }
  #pragma unroll
  for (int off = 16; off > 0; off >>= 1) acc += __shfl_xor(acc, off, 32);
  if (lane == 0) h2[wid] = acc;
}

// =====================================================================
// Layer-2 (scalar head) edge passes
// =====================================================================
__global__ void l2_edge_logits(const int* __restrict__ ei,
                               const float* __restrict__ h2,
                               const float* __restrict__ att_s,
                               const float* __restrict__ att_d,
                               float* __restrict__ e2,
                               unsigned* __restrict__ m2, int E, int Et) {
  const int e = blockIdx.x * blockDim.x + threadIdx.x;
  if (e >= Et) return;
  int s, d;
  if (e < E) { s = ei[e]; d = ei[E + e]; } else { s = d = e - E; }
  float v = att_s[0] * h2[s] + att_d[0] * h2[d];
  v = v > 0.f ? v : LEAKY * v;
  e2[e] = v;
  atomicMax(&m2[d], enc_f32(v));
}

__global__ void l2_edge_exp(const int* __restrict__ ei,
                            float* __restrict__ e2,
                            const unsigned* __restrict__ m2,
                            float* __restrict__ den2, int E, int Et) {
  const int e = blockIdx.x * blockDim.x + threadIdx.x;
  if (e >= Et) return;
  int d;
  if (e < E) { d = ei[E + e]; } else { d = e - E; }
  const float w = __expf(e2[e] - dec_f32(m2[d]));
  e2[e] = w;
  atomicAdd(&den2[d], w);
}

__global__ void l2_edge_agg(const int* __restrict__ ei,
                            const float* __restrict__ e2,
                            const float* __restrict__ den2,
                            const float* __restrict__ h2,
                            float* __restrict__ out, int E, int Et) {
  const int e = blockIdx.x * blockDim.x + threadIdx.x;
  if (e >= Et) return;
  int s, d;
  if (e < E) { s = ei[e]; d = ei[E + e]; } else { s = d = e - E; }
  const float alpha = e2[e] / den2[d];
  atomicAdd(&out[d], alpha * h2[s]);   // out pre-seeded with bias2
}

// =====================================================================
extern "C" void kernel_launch(void* const* d_in, const int* in_sizes, int n_in,
                              void* d_out, int out_size, void* d_ws, size_t ws_size,
                              hipStream_t stream) {
  const float* x        = (const float*)d_in[0];   // [N,128]
  const int*   ei       = (const int*)  d_in[1];   // [2,E]
  const float* W1       = (const float*)d_in[2];   // [128,256]
  const float* att_src1 = (const float*)d_in[3];   // [4,64]
  const float* att_dst1 = (const float*)d_in[4];   // [4,64]
  const float* bias1    = (const float*)d_in[5];   // [256]
  const float* W2       = (const float*)d_in[6];   // [256,1]
  const float* att_src2 = (const float*)d_in[7];   // [1,1]
  const float* att_dst2 = (const float*)d_in[8];   // [1,1]
  const float* bias2    = (const float*)d_in[9];   // [1]
  float* out = (float*)d_out;

  const int N  = in_sizes[0] / 128;
  const int E  = in_sizes[1] / 2;
  const int Et = N + E;                  // edges incl. self-loops

  // ---- carve workspace (256B-aligned) ----
  size_t off = 0;
  auto carve = [&](size_t bytes) -> char* {
    char* p = (char*)d_ws + off;
    off += (bytes + 255) & ~(size_t)255;
    return p;
  };
  float*    h    = (float*)   carve((size_t)N * 256 * 4);
  float*    asrc = (float*)   carve((size_t)N * 4 * 4);
  float*    adst = (float*)   carve((size_t)N * 4 * 4);
  unsigned* m1   = (unsigned*)carve((size_t)N * 4 * 4);
  float*    den1 = (float*)   carve((size_t)N * 4 * 4);
  float*    e1   = (float*)   carve((size_t)Et * 4 * 4);
  float*    agg1 = (float*)   carve((size_t)N * 256 * 4);
  float*    h2   = (float*)   carve((size_t)N * 4);
  unsigned* m2   = (unsigned*)carve((size_t)N * 4);
  float*    den2 = (float*)   carve((size_t)N * 4);
  float*    e2   = (float*)   carve((size_t)Et * 4);

  // 0) init accumulators, seed out with bias2
  init_ws<<<4096, 256, 0, stream>>>(m1, den1, agg1, m2, den2, out, bias2, N);

  // 1) h = x @ W1 (WMMA)
  {
    const long tilesM = (N + 15) / 16;
    const long waves  = tilesM * 16;               // 16 tiles along 256 cols
    const int  blocks = (int)((waves * 32 + 255) / 256);
    gemm_wmma<<<blocks, 256, 0, stream>>>(x, W1, h, N);
  }

  // 2) attention dot products
  {
    const int blocks = (int)(((long)N * 32 + 255) / 256);
    att_reduce<<<blocks, 256, 0, stream>>>(h, att_src1, att_dst1, asrc, adst, N);
  }

  // 3) layer-1 segment softmax + aggregation
  {
    const int blk_eh = (int)(((long)Et * 4 + 255) / 256);
    l1_edge_logits<<<blk_eh, 256, 0, stream>>>(ei, asrc, adst, e1, m1, E, Et);
    l1_edge_exp   <<<blk_eh, 256, 0, stream>>>(ei, e1, m1, den1, E, Et);
    l1_edge_agg   <<<(Et + 3) / 4, 256, 0, stream>>>(ei, e1, den1, h, agg1, E, Et);
  }

  // 4) bias + elu + matvec with W2
  {
    const int blocks = (int)(((long)N * 32 + 255) / 256);
    finalize_matvec<<<blocks, 256, 0, stream>>>(agg1, bias1, W2, h2, N);
  }

  // 5) layer-2 scalar softmax + aggregation into out
  {
    const int blk_e = (Et + 255) / 256;
    l2_edge_logits<<<blk_e, 256, 0, stream>>>(ei, h2, att_src2, att_dst2, e2, m2, E, Et);
    l2_edge_exp   <<<blk_e, 256, 0, stream>>>(ei, e2, m2, den2, E, Et);
    l2_edge_agg   <<<blk_e, 256, 0, stream>>>(ei, e2, den2, h2, out, E, Et);
  }
}
